// TinyTransformer_28733331210335
// MI455X (gfx1250) — matrix operand order repaired and checked by reading, not compiled, not run on hardware
//
#include <hip/hip_runtime.h>
#include <hip/hip_bf16.h>

typedef __bf16 bf16;
typedef __attribute__((ext_vector_type(16))) __bf16 v16bf;
typedef __attribute__((ext_vector_type(8)))  float  v8f;

#define BATCH 64
#define SEQ   2048
#define DIM   32
#define VOCAB 32000

#if defined(__has_builtin)
#if __has_builtin(__builtin_amdgcn_global_load_async_to_lds_b128)
#define HAVE_ASYNC_LDS 1
#endif
#endif
#ifndef HAVE_ASYNC_LDS
#define HAVE_ASYNC_LDS 0
#endif

#if defined(__has_builtin)
#if __has_builtin(__builtin_amdgcn_s_wait_asynccnt)
#define WAIT_ASYNC() __builtin_amdgcn_s_wait_asynccnt(0)
#else
#define WAIT_ASYNC() asm volatile("s_wait_asynccnt 0x0" ::: "memory")
#endif
#else
#define WAIT_ASYNC() asm volatile("s_wait_asynccnt 0x0" ::: "memory")
#endif

// Exact pointer types for the async builtin (from hipcc diagnostic):
//   param0: int __vector(4) * in addrspace(1) (__device__/global)
//   param1: LDS-side int __vector(4) * (addrspace(3))
typedef int v4i __attribute__((vector_size(16)));
typedef __attribute__((address_space(1))) v4i as1_v4i;
typedef __attribute__((address_space(3))) v4i as3_v4i;

// ---------------------------------------------------------------------------
// Fragment loaders (CDNA5 WMMA 16x16x32 bf16 layouts, ISA 7.12.2)
// ---------------------------------------------------------------------------

// A-matrix 16x32 bf16, row-major source (ld = row stride in elements).
// lane: m = lane&15, lh = lane>>4; elems 0..7 -> K=8*lh+e ; 8..15 -> K=16+8*lh+(e-8)
__device__ inline v16bf load_a_rowmajor(const bf16* base, int ld, int lane) {
  int m  = lane & 15;
  int lh = lane >> 4;
  const bf16* row = base + (long)m * ld;
  union { uint4 u[2]; v16bf v; } cv;
  cv.u[0] = *(const uint4*)(row + 8 * lh);        // K = 8*lh .. 8*lh+7
  cv.u[1] = *(const uint4*)(row + 16 + 8 * lh);   // K = 16+8*lh .. +7
  return cv.v;
}

// Loads 16 contiguous bf16 of one K row (elements e = N); WMMA B holds N = lane&15, k in elements.
__device__ inline v16bf load_b_contig(const bf16* krow_n0) {
  union { uint4 u[2]; v16bf v; } cv;
  cv.u[0] = *(const uint4*)(krow_n0);
  cv.u[1] = *(const uint4*)(krow_n0 + 8);
  return cv.v;
}

// ---------------------------------------------------------------------------
// Kernel 1: embedding gather + Q/K/V projection, bf16 outputs.
// Q stored transposed [b][d][s] (pre-scaled by 1/sqrt(D)); K,V row-major [b][s][d].
// ---------------------------------------------------------------------------
__global__ __launch_bounds__(256) void qkv_embed_kernel(
    const int* __restrict__ x, const float* __restrict__ embed,
    const float* __restrict__ wq, const float* __restrict__ wk,
    const float* __restrict__ wv,
    bf16* __restrict__ Qt, bf16* __restrict__ Kr, bf16* __restrict__ Vr) {
  __shared__ float w[3 * DIM * DIM];
  int tid = threadIdx.x;
  for (int i = tid; i < DIM * DIM; i += 256) {
    w[i]                 = wq[i];
    w[DIM * DIM + i]     = wk[i];
    w[2 * DIM * DIM + i] = wv[i];
  }
  __syncthreads();

  long idx = (long)blockIdx.x * 256 + tid;   // row in [0, BATCH*SEQ)
  int b = (int)(idx / SEQ);
  int s = (int)(idx % SEQ);
  int tok = x[idx];
  const float* h = embed + (long)tok * DIM;
  float hr[DIM];
#pragma unroll
  for (int d = 0; d < DIM; ++d) hr[d] = h[d];

  const float qscale = 0.17677669529663687f;  // 1/sqrt(32) folded into Q
#pragma unroll 4
  for (int e = 0; e < DIM; ++e) {
    float q = 0.f, k = 0.f, v = 0.f;
#pragma unroll
    for (int d = 0; d < DIM; ++d) {
      q = fmaf(hr[d], w[e * DIM + d], q);
      k = fmaf(hr[d], w[DIM * DIM + e * DIM + d], k);
      v = fmaf(hr[d], w[2 * DIM * DIM + e * DIM + d], v);
    }
    Qt[((long)b * DIM + e) * SEQ + s] = (bf16)(q * qscale);
    Kr[((long)b * SEQ + s) * DIM + e] = (bf16)k;
    Vr[((long)b * SEQ + s) * DIM + e] = (bf16)v;
  }
}

// ---------------------------------------------------------------------------
// Kernel 2: flash attention, 8 waves/block, 16 queries/wave, key tiles of 32.
// Scores computed TRANSPOSED (A = K-tile, B = Q^T) so that score C-fragments
// are already in P's A-fragment layout for the P x V WMMA (no lane shuffle).
// K/V tiles are double-buffered in LDS via GLOBAL_LOAD_ASYNC_TO_LDS_B128
// (ASYNCcnt) when available: tile t+1 DMA overlaps tile t compute.
// ---------------------------------------------------------------------------
__global__ __launch_bounds__(256) void flash_attn_kernel(
    const bf16* __restrict__ Qt, const bf16* __restrict__ Kr,
    const bf16* __restrict__ Vr, float* __restrict__ pooledPart) {
  __shared__ float wsum[8][32];

  int b    = blockIdx.x >> 4;        // 16 query blocks per batch
  int qblk = blockIdx.x & 15;
  int wave = threadIdx.x >> 5;
  int lane = threadIdx.x & 31;
  int lh   = lane >> 4;              // lane half
  int q0   = qblk * 128 + wave * 16; // this wave's 16 queries

  // B-fragment of Q^T: lane&15 = query (N), elements = d in table order. Reused all tiles.
  v16bf bq;
  {
    const bf16* qp = Qt + (long)b * DIM * SEQ + q0 + (lane & 15);
#pragma unroll
    for (int e = 0; e < 16; ++e) {
      int kd = (e < 8) ? (8 * lh + e) : (16 + 8 * lh + (e - 8));
      bq[e] = qp[(long)kd * SEQ];
    }
  }

  const bf16* Kb = Kr + (long)b * SEQ * DIM;
  const bf16* Vb = Vr + (long)b * SEQ * DIM;

#if HAVE_ASYNC_LDS
  __shared__ bf16 ktile[2][32 * DIM];   // 2 x 2KB
  __shared__ bf16 vtile[2][32 * DIM];   // 2 x 2KB
  const int tid = threadIdx.x;
  // Cooperative tile copy: 256 threads x b128 = 4KB = one K tile + one V tile.
  auto issue_copy = [&](int k0, int buf) {
    int c = tid & 127;                                  // 16B chunk index
    const bf16* gsrc = ((tid < 128) ? Kb : Vb) + (long)k0 * DIM + c * 8;
    bf16* ldst = ((tid < 128) ? &ktile[buf][0] : &vtile[buf][0]) + c * 8;
    __builtin_amdgcn_global_load_async_to_lds_b128(
        (as1_v4i*)(uintptr_t)gsrc,
        (as3_v4i*)(unsigned)(uintptr_t)ldst,
        /*offset=*/0, /*cpol=*/0);
  };
  // A-fragment from an LDS tile (16 rows, row stride DIM) -> ds_load_b128 x2
  auto load_a_lds = [&](const bf16* base) -> v16bf {
    int m = lane & 15;
    const bf16* row = base + m * DIM;
    union { uint4 u[2]; v16bf v; } cv;
    cv.u[0] = *(const uint4*)(row + 8 * lh);
    cv.u[1] = *(const uint4*)(row + 16 + 8 * lh);
    return cv.v;
  };
  issue_copy(0, 0);   // prime the pipeline with tile 0
#endif

  float m_run = -1e30f, l_run = 0.f;
  v8f o1 = {};  // context accumulator, channels 0..15  (C layout: N=lane&15, M=8*lh+r)
  v8f o2 = {};  // channels 16..31

  for (int t = 0; t < SEQ / 32; ++t) {
    int k0 = t * 32;
#if HAVE_ASYNC_LDS
    int cur = t & 1;
    WAIT_ASYNC();        // this wave's copy of tile t has landed
    __syncthreads();     // all waves' copies landed; all done reading buf cur^1
    if (t + 1 < SEQ / 32) issue_copy(k0 + 32, cur ^ 1);  // overlap DMA w/ compute
    v16bf ak1 = load_a_lds(&ktile[cur][0]);          // keys k0..k0+15
    v16bf ak2 = load_a_lds(&ktile[cur][16 * DIM]);   // keys k0+16..k0+31
#else
    if (k0 + 32 < SEQ) {  // prefetch next key/value tile (global_prefetch_b8)
      __builtin_prefetch(Kb + (long)(k0 + 32 + lane) * DIM, 0, 3);
      __builtin_prefetch(Vb + (long)(k0 + 32 + lane) * DIM, 0, 3);
    }
    v16bf ak1 = load_a_rowmajor(Kb + (long)k0 * DIM, DIM, lane);
    v16bf ak2 = load_a_rowmajor(Kb + (long)(k0 + 16) * DIM, DIM, lane);
#endif

    // --- scores^T = K_tile x Q^T ------------------------------------------
    v8f c1 = {}, c2 = {};
    c1 = __builtin_amdgcn_wmma_f32_16x16x32_bf16(false, ak1, false, bq,
                                                 (short)0, c1, false, false);
    c2 = __builtin_amdgcn_wmma_f32_16x16x32_bf16(false, ak2, false, bq,
                                                 (short)0, c2, false, false);
    // lane holds query q=lane&15; c1[r] = key 8*lh+r, c2[r] = key 16+8*lh+r

    // --- online softmax (fp32 stats) --------------------------------------
    float mt = -1e30f;
#pragma unroll
    for (int r = 0; r < 8; ++r) { mt = fmaxf(mt, c1[r]); mt = fmaxf(mt, c2[r]); }
    mt = fmaxf(mt, __shfl_xor(mt, 16, 32));       // join the two key halves
    float m_new = fmaxf(m_run, mt);
    float alpha = __expf(m_run - m_new);

    float p1[8], p2[8], lt = 0.f;
#pragma unroll
    for (int r = 0; r < 8; ++r) {
      p1[r] = __expf(c1[r] - m_new);
      p2[r] = __expf(c2[r] - m_new);
      lt += p1[r] + p2[r];
    }
    lt += __shfl_xor(lt, 16, 32);
    l_run = l_run * alpha + lt;
    m_run = m_new;

    // --- pack P directly into A-fragment layout (no cross-lane moves) -----
    v16bf pa;
#pragma unroll
    for (int r = 0; r < 8; ++r) {
      pa[r]     = (bf16)p1[r];   // K(e)=8*lh+e        == c1's key
      pa[8 + r] = (bf16)p2[r];   // K(e)=16+8*lh+(e-8) == c2's key
    }

    // --- rescale running context (query of (lane,r) is 8*lh+r) ------------
#pragma unroll
    for (int r = 0; r < 8; ++r) {
      float sc = __shfl(alpha, 8 * lh + r, 32);
      o1[r] *= sc;
      o2[r] *= sc;
    }

    // --- context += P x V --------------------------------------------------
#if HAVE_ASYNC_LDS
    const bf16* vrow = &vtile[cur][lane & 15];       // lane&15 = channel (N) of B
#else
    const bf16* vrow = Vb + (long)k0 * DIM + (lane & 15);  // lane&15 = channel (N) of B
#endif
    v16bf bv1, bv2;
#pragma unroll
    for (int e = 0; e < 16; ++e) {
      int kk = (e < 8) ? (8 * lh + e) : (16 + 8 * lh + (e - 8));
      bv1[e] = vrow[kk * DIM];        // channels 0..15
      bv2[e] = vrow[kk * DIM + 16];   // channels 16..31
    }
    o1 = __builtin_amdgcn_wmma_f32_16x16x32_bf16(false, pa, false, bv1,
                                                 (short)0, o1, false, false);
    o2 = __builtin_amdgcn_wmma_f32_16x16x32_bf16(false, pa, false, bv2,
                                                 (short)0, o2, false, false);
  }

  // --- finalize: divide by softmax denom, sum over this wave's 16 queries ---
  float s1 = 0.f, s2 = 0.f;
#pragma unroll
  for (int r = 0; r < 8; ++r) {
    float denom = __shfl(l_run, 8 * lh + r, 32);
    float inv = 1.0f / denom;
    s1 += o1[r] * inv;
    s2 += o2[r] * inv;
  }
  s1 += __shfl_xor(s1, 16, 32);  // both query halves
  s2 += __shfl_xor(s2, 16, 32);
  // channel for o1 = lane&15, for o2 = 16+(lane&15)  ==> lane writes channel `lane`
  wsum[wave][lane] = (lane < 16) ? s1 : s2;
  __syncthreads();

  if (threadIdx.x < 32) {
    float tot = 0.f;
#pragma unroll
    for (int w = 0; w < 8; ++w) tot += wsum[w][threadIdx.x];
    pooledPart[((long)b * 16 + qblk) * 32 + threadIdx.x] = tot;
  }
}

// ---------------------------------------------------------------------------
// Kernel 3: pooled = (mean_s context) @ out_w^T + out_b ;
//           logits = pooled @ vocab_w^T + vocab_b
// grid (125, 64): each block = 256 vocab rows of one batch element.
// ---------------------------------------------------------------------------
__global__ __launch_bounds__(256) void logits_kernel(
    const float* __restrict__ pooledPart, const float* __restrict__ out_w,
    const float* __restrict__ out_b, const float* __restrict__ vocab_w,
    const float* __restrict__ vocab_b, float* __restrict__ out) {
  __shared__ float ctx[32];
  __shared__ float pooled[32];
  int b = blockIdx.y;
  int v = blockIdx.x * 256 + threadIdx.x;

  if (threadIdx.x < 32) {
    float a = 0.f;
#pragma unroll
    for (int j = 0; j < 16; ++j)
      a += pooledPart[((long)b * 16 + j) * 32 + threadIdx.x];
    ctx[threadIdx.x] = a * (1.0f / SEQ);   // mean over sequence
  }
  __syncthreads();
  if (threadIdx.x < 32) {
    float a = out_b[threadIdx.x];
#pragma unroll
    for (int d = 0; d < 32; ++d)
      a = fmaf(ctx[d], out_w[threadIdx.x * 32 + d], a);
    pooled[threadIdx.x] = a;
  }
  __syncthreads();

  float acc = vocab_b[v];
  const float* wr = vocab_w + (long)v * 32;
#pragma unroll
  for (int e = 0; e < 32; ++e) acc = fmaf(pooled[e], wr[e], acc);
  out[(long)b * VOCAB + v] = acc;
}

// ---------------------------------------------------------------------------
extern "C" void kernel_launch(void* const* d_in, const int* in_sizes, int n_in,
                              void* d_out, int out_size, void* d_ws, size_t ws_size,
                              hipStream_t stream) {
  const int*   x       = (const int*)d_in[0];
  const float* embed   = (const float*)d_in[1];
  const float* wq      = (const float*)d_in[2];
  const float* wk      = (const float*)d_in[3];
  const float* wv      = (const float*)d_in[4];
  const float* out_w   = (const float*)d_in[5];
  const float* out_b   = (const float*)d_in[6];
  const float* vocab_w = (const float*)d_in[7];
  const float* vocab_b = (const float*)d_in[8];
  float* out = (float*)d_out;

  // workspace layout: Qt | K | V (bf16, 8 MB each) | pooled partials (f32, 128 KB)
  bf16* Qt = (bf16*)d_ws;
  bf16* Kr = Qt + (size_t)BATCH * DIM * SEQ;
  bf16* Vr = Kr + (size_t)BATCH * SEQ * DIM;
  float* pooledPart = (float*)(Vr + (size_t)BATCH * SEQ * DIM);

  qkv_embed_kernel<<<(BATCH * SEQ) / 256, 256, 0, stream>>>(
      x, embed, wq, wk, wv, Qt, Kr, Vr);
  flash_attn_kernel<<<BATCH * (SEQ / 128), 256, 0, stream>>>(
      Qt, Kr, Vr, pooledPart);
  logits_kernel<<<dim3(VOCAB / 256, BATCH), 256, 0, stream>>>(
      pooledPart, out_w, out_b, vocab_w, vocab_b, out);
}
